// _ATTR_NETWORK_2697239462734
// MI455X (gfx1250) — compile-verified, matrix operand order checked by reading, not run on hardware
//
#include <hip/hip_runtime.h>
#include <hip/hip_bf16.h>
#include <cstdint>
#include <cstddef>

// ---------------------------------------------------------------------------
// CDNA5 (gfx1250) WMMA types / helpers
// ---------------------------------------------------------------------------
typedef __attribute__((ext_vector_type(16))) __bf16        v16bf;
typedef __attribute__((ext_vector_type(8)))  float         v8f;

#define DEV __device__ __forceinline__

// f32 -> bf16 bits, round-to-nearest-even
DEV unsigned short f2bf(float f) {
    unsigned int a = __builtin_bit_cast(unsigned int, f);
    a = (a + 0x7FFFu + ((a >> 16) & 1u)) >> 16;
    return (unsigned short)a;
}

// Load a 16x32 bf16 A/B fragment from a bf16 row (row-major, this lane's row).
// Per ISA 7.12.2 (16-bit A 16x32): lanes 0-15 hold K {0-7,16-23}, lanes 16-31
// hold K {8-15,24-31}; B operand uses the identical layout on W stored [N,K].
// Two 16-byte loads, no conversion VALU.
struct U32x8 { uint4 a, b; };
DEV v16bf load_frag(const unsigned short* __restrict__ p, int lane) {
    p += (lane & 16) ? 8 : 0;
    U32x8 t;
    t.a = *(const uint4*)(p);
    t.b = *(const uint4*)(p + 16);
    return __builtin_bit_cast(v16bf, t);
}

DEV v8f wmma_bf16(v16bf a, v16bf b, v8f c) {
    return __builtin_amdgcn_wmma_f32_16x16x32_bf16(
        /*neg_a=*/false, a, /*neg_b=*/false, b,
        /*c_mod=*/(short)0, c, /*reuse_a=*/false, /*reuse_b=*/false);
}

// ---------------------------------------------------------------------------
// Problem constants
// ---------------------------------------------------------------------------
#define BB   512
#define LL   192
#define DD   256
#define FFD  1024
#define NH   4
#define DH   64
#define NT   330   // 30 pos + 300 neg
#define NEGV (-1e9f)

typedef unsigned short bfu;   // bf16 bit pattern

// ---------------------------------------------------------------------------
// 0) Elementwise f32 -> bf16 (weight conversion, once per launch)
// ---------------------------------------------------------------------------
__global__ __launch_bounds__(256)
void k_f2bf(const float* __restrict__ src, bfu* __restrict__ dst) {
    size_t i = (size_t)blockIdx.x * 256 + threadIdx.x;
    dst[i] = f2bf(src[i]);
}

// ---------------------------------------------------------------------------
// 1) Embedding gather: x = attr_emb[attr] + ind_emb[attr_inds]  (f32 + bf16)
// ---------------------------------------------------------------------------
__global__ __launch_bounds__(256)
void k_embed(const int* __restrict__ attr, const int* __restrict__ inds,
             const float* __restrict__ aemb, const float* __restrict__ iemb,
             float* __restrict__ x, bfu* __restrict__ xb) {
    size_t idx = (size_t)blockIdx.x * 256 + threadIdx.x;   // B*L*D threads
    size_t bl  = idx >> 8;
    int    d   = (int)(idx & 255);
    float v = aemb[(size_t)attr[bl] * DD + d] + iemb[(size_t)inds[bl] * DD + d];
    x[idx]  = v;
    xb[idx] = f2bf(v);
}

// ---------------------------------------------------------------------------
// 2) WMMA GEMM:  Y[M,N] = A[M,K] @ W[N,K]^T (+bias)(+relu), A/W bf16, acc f32
//    block = 128 threads (4 waves), block tile 64M x 64N, wave tile 16M x 64N
//    Software-pipelined k-loop: next-step fragments are in flight while the
//    current step's WMMAs execute (avoids s_wait_loadcnt 0 before each WMMA).
//    flags: 1=bias, 2=relu, 4=store bf16 (else f32)
// ---------------------------------------------------------------------------
__global__ __launch_bounds__(128)
void k_gemm(const bfu* __restrict__ A, const bfu* __restrict__ W,
            const float* __restrict__ bias, void* __restrict__ Y,
            int N, int K, int flags) {
    const int lane = threadIdx.x & 31;
    const int wave = threadIdx.x >> 5;
    const int n0   = blockIdx.x * 64;
    const int m0   = blockIdx.y * 64 + wave * 16;

    const bfu* ap  = A + (size_t)(m0 + (lane & 15)) * K;
    const bfu* wp0 = W + (size_t)(n0 + (lane & 15)) * K;
    const bfu* wp1 = wp0 + (size_t)16 * K;
    const bfu* wp2 = wp0 + (size_t)32 * K;
    const bfu* wp3 = wp0 + (size_t)48 * K;

    v8f acc[4] = {};

    // prologue: fragments for k = 0
    v16bf a_c  = load_frag(ap,  lane);
    v16bf b_c0 = load_frag(wp0, lane);
    v16bf b_c1 = load_frag(wp1, lane);
    v16bf b_c2 = load_frag(wp2, lane);
    v16bf b_c3 = load_frag(wp3, lane);

    for (int k = 32; k < K; k += 32) {
        // prefetch next step while current WMMAs run
        v16bf a_n  = load_frag(ap  + k, lane);
        v16bf b_n0 = load_frag(wp0 + k, lane);
        v16bf b_n1 = load_frag(wp1 + k, lane);
        v16bf b_n2 = load_frag(wp2 + k, lane);
        v16bf b_n3 = load_frag(wp3 + k, lane);

        acc[0] = wmma_bf16(a_c, b_c0, acc[0]);
        acc[1] = wmma_bf16(a_c, b_c1, acc[1]);
        acc[2] = wmma_bf16(a_c, b_c2, acc[2]);
        acc[3] = wmma_bf16(a_c, b_c3, acc[3]);

        a_c = a_n; b_c0 = b_n0; b_c1 = b_n1; b_c2 = b_n2; b_c3 = b_n3;
    }
    acc[0] = wmma_bf16(a_c, b_c0, acc[0]);
    acc[1] = wmma_bf16(a_c, b_c1, acc[1]);
    acc[2] = wmma_bf16(a_c, b_c2, acc[2]);
    acc[3] = wmma_bf16(a_c, b_c3, acc[3]);

    const int mrow = m0 + ((lane & 16) ? 8 : 0);
#pragma unroll
    for (int r = 0; r < 8; ++r) {
        const size_t rowoff = (size_t)(mrow + r) * N;
#pragma unroll
        for (int j = 0; j < 4; ++j) {
            int n = n0 + j * 16 + (lane & 15);
            float v = acc[j][r];
            if (flags & 1) v += bias[n];
            if (flags & 2) v = fmaxf(v, 0.0f);
            if (flags & 4) ((bfu*)Y)[rowoff + n]   = f2bf(v);
            else           ((float*)Y)[rowoff + n] = v;
        }
    }
}

// ---------------------------------------------------------------------------
// 3) Repack qkv(bf16) [B*L,768] -> qh,kh [B,H,L,DH] and vT [B,H,DH,L] (bf16)
// ---------------------------------------------------------------------------
__global__ __launch_bounds__(256)
void k_repack(const bfu* __restrict__ qkv, bfu* __restrict__ qh,
              bfu* __restrict__ kh, bfu* __restrict__ vT) {
    size_t idx = (size_t)blockIdx.x * 256 + threadIdx.x;   // B*L*D threads
    size_t bl  = idx >> 8;
    int    d   = (int)(idx & 255);
    int b = (int)(bl / LL), l = (int)(bl % LL);
    int h = d >> 6, dh = d & 63;
    size_t bh = (size_t)b * NH + h;
    const bfu* src = qkv + bl * (3 * DD);
    qh[(bh * LL + l) * DH + dh] = src[d];
    kh[(bh * LL + l) * DH + dh] = src[DD + d];
    vT[(bh * DH + dh) * LL + l] = src[2 * DD + d];
}

// ---------------------------------------------------------------------------
// 4) Fused attention per (b,h,q-tile): WMMA scores -> masked softmax -> P@V^T
//    One wave per block. P staged through LDS (bf16) to re-fragment C->A layout.
//    Both WMMA loops are software-pipelined.
// ---------------------------------------------------------------------------
__global__ __launch_bounds__(32)
void k_attn(const bfu* __restrict__ qh, const bfu* __restrict__ kh,
            const bfu* __restrict__ vT, const int* __restrict__ lens,
            bfu* __restrict__ ctx) {
    const int qt   = blockIdx.x % (LL / 16);
    const int bh   = blockIdx.x / (LL / 16);
    const int h    = bh & (NH - 1);
    const int b    = bh >> 2;
    const int lane = threadIdx.x;
    const int len  = lens[b];

    // ---- scores = Q K^T (pipelined over key tiles) ---------------------
    const bfu* qbase = qh + ((size_t)bh * LL + qt * 16 + (lane & 15)) * DH;
    const v16bf qa0 = load_frag(qbase,      lane);
    const v16bf qa1 = load_frag(qbase + 32, lane);

    const bfu* kbase = kh + ((size_t)bh * LL + (lane & 15)) * DH;
    v8f sc[LL / 16] = {};
    v16bf kb0 = load_frag(kbase,      lane);
    v16bf kb1 = load_frag(kbase + 32, lane);
#pragma unroll
    for (int jt = 0; jt < LL / 16; ++jt) {
        v16bf nx0 = kb0, nx1 = kb1;
        if (jt + 1 < LL / 16) {
            const bfu* nb = kbase + (size_t)(jt + 1) * 16 * DH;
            nx0 = load_frag(nb,      lane);
            nx1 = load_frag(nb + 32, lane);
        }
        sc[jt] = wmma_bf16(qa0, kb0, sc[jt]);
        sc[jt] = wmma_bf16(qa1, kb1, sc[jt]);
        kb0 = nx0; kb1 = nx1;
    }

    // ---- scale + key mask ---------------------------------------------
    const float scale = 0.125f;   // 1/sqrt(64)
#pragma unroll
    for (int jt = 0; jt < LL / 16; ++jt) {
        bool valid = (jt * 16 + (lane & 15)) < len;
#pragma unroll
        for (int r = 0; r < 8; ++r)
            sc[jt][r] = valid ? sc[jt][r] * scale : NEGV;
    }

    // ---- row softmax: C-layout row = r + (lane>=16?8:0); reduce over 16 lanes
    float inv[8];
#pragma unroll
    for (int r = 0; r < 8; ++r) {
        float mx = NEGV;
#pragma unroll
        for (int jt = 0; jt < LL / 16; ++jt) mx = fmaxf(mx, sc[jt][r]);
#pragma unroll
        for (int off = 8; off > 0; off >>= 1) mx = fmaxf(mx, __shfl_xor(mx, off, 16));
        float s = 0.0f;
#pragma unroll
        for (int jt = 0; jt < LL / 16; ++jt) {
            float e = __expf(sc[jt][r] - mx);
            sc[jt][r] = e;
            s += e;
        }
#pragma unroll
        for (int off = 8; off > 0; off >>= 1) s += __shfl_xor(s, off, 16);
        inv[r] = 1.0f / s;
    }

    // ---- P: C-layout -> LDS row-major bf16, re-read in A-layout --------
    __shared__ __align__(16) bfu P[16][LL];
#pragma unroll
    for (int jt = 0; jt < LL / 16; ++jt)
#pragma unroll
        for (int r = 0; r < 8; ++r)
            P[r + ((lane & 16) ? 8 : 0)][jt * 16 + (lane & 15)] = f2bf(sc[jt][r] * inv[r]);
    __syncthreads();

    // ---- ctx = P @ V^T (pipelined over K tiles) ------------------------
    const bfu* prow  = &P[lane & 15][0];
    const bfu* vbase = vT + ((size_t)bh * DH + (lane & 15)) * LL;
    v8f acc[4] = {};
    v16bf pa = load_frag(prow, lane);
    v16bf vb0 = load_frag(vbase,               lane);
    v16bf vb1 = load_frag(vbase + (size_t)16 * LL, lane);
    v16bf vb2 = load_frag(vbase + (size_t)32 * LL, lane);
    v16bf vb3 = load_frag(vbase + (size_t)48 * LL, lane);
#pragma unroll
    for (int kt = 0; kt < LL / 32; ++kt) {
        v16bf pn = pa, n0 = vb0, n1 = vb1, n2 = vb2, n3 = vb3;
        if (kt + 1 < LL / 32) {
            int k = (kt + 1) * 32;
            pn = load_frag(prow + k, lane);
            n0 = load_frag(vbase + k,                  lane);
            n1 = load_frag(vbase + (size_t)16 * LL + k, lane);
            n2 = load_frag(vbase + (size_t)32 * LL + k, lane);
            n3 = load_frag(vbase + (size_t)48 * LL + k, lane);
        }
        acc[0] = wmma_bf16(pa, vb0, acc[0]);
        acc[1] = wmma_bf16(pa, vb1, acc[1]);
        acc[2] = wmma_bf16(pa, vb2, acc[2]);
        acc[3] = wmma_bf16(pa, vb3, acc[3]);
        pa = pn; vb0 = n0; vb1 = n1; vb2 = n2; vb3 = n3;
    }

    // ---- write ctx (bf16) [B,L,D], d = h*64 + jn*16 + lane%16 ----------
    const int mbase = qt * 16 + ((lane & 16) ? 8 : 0);
#pragma unroll
    for (int jn = 0; jn < 4; ++jn)
#pragma unroll
        for (int r = 0; r < 8; ++r) {
            int l = mbase + r;
            ctx[((size_t)b * LL + l) * DD + h * DH + jn * 16 + (lane & 15)] = f2bf(acc[jn][r]);
        }
}

// ---------------------------------------------------------------------------
// 5) Residual + LayerNorm over D=256; one wave (32 lanes x 8 elems) per row.
//    Writes f32 (residual / pooling) and bf16 (next GEMM A operand).
// ---------------------------------------------------------------------------
__global__ __launch_bounds__(256)
void k_add_ln(const float* __restrict__ X, const float* __restrict__ Yin,
              const float* __restrict__ s, const float* __restrict__ bp,
              float* __restrict__ Xout, bfu* __restrict__ Xbf) {
    const int lane = threadIdx.x & 31;
    const int wave = threadIdx.x >> 5;
    const size_t row = (size_t)blockIdx.x * 8 + wave;
    const float* xr = X   + row * DD;
    const float* yr = Yin + row * DD;
    float t[8];
    float sum = 0.0f;
#pragma unroll
    for (int i = 0; i < 8; ++i) { t[i] = xr[lane + i * 32] + yr[lane + i * 32]; sum += t[i]; }
#pragma unroll
    for (int off = 16; off > 0; off >>= 1) sum += __shfl_xor(sum, off);
    const float mean = sum * (1.0f / DD);
    float vs = 0.0f;
#pragma unroll
    for (int i = 0; i < 8; ++i) { float dv = t[i] - mean; vs += dv * dv; }
#pragma unroll
    for (int off = 16; off > 0; off >>= 1) vs += __shfl_xor(vs, off);
    const float rstd = rsqrtf(vs * (1.0f / DD) + 1e-5f);
    float* o  = Xout + row * DD;
    bfu*   ob = Xbf  + row * DD;
#pragma unroll
    for (int i = 0; i < 8; ++i) {
        int d = lane + i * 32;
        float v = (t[i] - mean) * rstd * s[d] + bp[d];
        o[d]  = v;
        ob[d] = f2bf(v);
    }
}

// ---------------------------------------------------------------------------
// 6) Ragged weighted pooling + concat with user/item embedding gathers
// ---------------------------------------------------------------------------
__global__ __launch_bounds__(256)
void k_pool(const float* __restrict__ x, const float* __restrict__ attr_tf,
            const float* __restrict__ attr_feat,
            const int* __restrict__ lens_u, const int* __restrict__ lens_i,
            const int* __restrict__ user_ids, const int* __restrict__ item_ids,
            const float* __restrict__ uemb, const float* __restrict__ iemb,
            const float* __restrict__ fw, const float* __restrict__ fb,
            float* __restrict__ uout, float* __restrict__ iout) {
    const int b = blockIdx.x, d = threadIdx.x;
    __shared__ float wl[LL], tf[LL];
    if (d < LL) {
        const float* f = attr_feat + ((size_t)b * LL + d) * 13;
        float a = fb[0];
#pragma unroll
        for (int j = 0; j < 13; ++j) a += f[j] * fw[j];
        wl[d] = 1.0f / (1.0f + __expf(-a));
        tf[d] = attr_tf[(size_t)b * LL + d];
    }
    __syncthreads();
    const int Lu = lens_u[b], Li = lens_i[b];
    const int Lt = Lu + Li;
    float su = 0.0f, si = 0.0f;
    const float* xb = x + (size_t)b * LL * DD + d;
    for (int l = 0; l < Lt; ++l) {
        float xv = xb[(size_t)l * DD] * tf[l];
        if (l < Lu) su += xv * wl[l];
        else        si += xv * (1.0f - wl[l]);
    }
    uout[(size_t)b * 512 + d]       = su / (float)Lu;
    iout[(size_t)b * 512 + d]       = si / (float)Li;
    uout[(size_t)b * 512 + 256 + d] = uemb[(size_t)user_ids[b] * DD + d];
    iout[(size_t)b * 512 + 256 + d] = iemb[(size_t)item_ids[b] * DD + d];
}

// ---------------------------------------------------------------------------
// 7) Logit head (bandwidth-bound gather-dot): one wave per (b, target)
//    out = [logits | mask | new_targets] each [B, 330], as f32
// ---------------------------------------------------------------------------
__global__ __launch_bounds__(256)
void k_logits(const int* __restrict__ pos_t, const int* __restrict__ pos_l,
              const int* __restrict__ neg_t, const int* __restrict__ neg_l,
              const float* __restrict__ oeu, const float* __restrict__ oei,
              const float* __restrict__ uout, const float* __restrict__ iout,
              float* __restrict__ out) {
    const int lane = threadIdx.x & 31;
    const int wave = threadIdx.x >> 5;
    const int pair = blockIdx.x * 8 + wave;          // < B*NT exactly
    const int b  = pair / NT;
    const int kk = pair % NT;
    int tgt, valid;
    if (kk < 30) { tgt = pos_t[b * 30 + kk];        valid = kk < pos_l[b]; }
    else         { tgt = neg_t[b * 300 + (kk - 30)]; valid = (kk - 30) < neg_l[b]; }
    const float* eu = oeu + (size_t)tgt * 512;
    const float* ei = oei + (size_t)tgt * 512;
    const float* u  = uout + (size_t)b * 512;
    const float* it = iout + (size_t)b * 512;
    float sum = 0.0f;
#pragma unroll
    for (int j = lane; j < 512; j += 32) sum += eu[j] * u[j] + ei[j] * it[j];
#pragma unroll
    for (int off = 16; off > 0; off >>= 1) sum += __shfl_xor(sum, off);
    if (lane == 0) {
        const size_t T = (size_t)BB * NT;
        out[pair]         = sum;
        out[T + pair]     = valid ? 1.0f : 0.0f;
        out[2 * T + pair] = (kk < 30 && valid) ? 1.0f : 0.0f;
    }
}

// ---------------------------------------------------------------------------
// Host launcher
// ---------------------------------------------------------------------------
extern "C" void kernel_launch(void* const* d_in, const int* in_sizes, int n_in,
                              void* d_out, int out_size, void* d_ws, size_t ws_size,
                              hipStream_t stream) {
    (void)in_sizes; (void)n_in; (void)out_size; (void)ws_size;

    const int*   attr       = (const int*)  d_in[0];
    const int*   attr_inds  = (const int*)  d_in[1];
    const float* attr_tf    = (const float*)d_in[2];
    const float* attr_feat  = (const float*)d_in[3];
    const int*   attr_lens  = (const int*)  d_in[4];
    const int*   lens_u     = (const int*)  d_in[5];
    const int*   lens_i     = (const int*)  d_in[6];
    const int*   user_ids   = (const int*)  d_in[7];
    const int*   item_ids   = (const int*)  d_in[8];
    const int*   pos_t      = (const int*)  d_in[9];
    const int*   pos_l      = (const int*)  d_in[10];
    const int*   neg_t      = (const int*)  d_in[11];
    const int*   neg_l      = (const int*)  d_in[12];
    const float* attr_emb   = (const float*)d_in[13];
    const float* ind_emb    = (const float*)d_in[14];
    const float* uemb       = (const float*)d_in[15];
    const float* iemb       = (const float*)d_in[16];
    const float* oeu        = (const float*)d_in[17];
    const float* oei        = (const float*)d_in[18];
    const float* feat_w     = (const float*)d_in[19];
    const float* feat_b     = (const float*)d_in[20];
    const float* Wqkv       = (const float*)d_in[21];
    const float* bqkv       = (const float*)d_in[22];
    const float* Wo         = (const float*)d_in[23];
    const float* bo         = (const float*)d_in[24];
    const float* ln1_s      = (const float*)d_in[25];
    const float* ln1_b      = (const float*)d_in[26];
    const float* W1         = (const float*)d_in[27];
    const float* b1         = (const float*)d_in[28];
    const float* W2         = (const float*)d_in[29];
    const float* b2         = (const float*)d_in[30];
    const float* ln2_s      = (const float*)d_in[31];
    const float* ln2_b      = (const float*)d_in[32];

    const int M = BB * LL;                              // 98304 rows
    const size_t S = (size_t)M * DD * sizeof(float);    // 100,663,296 B

    char* ws = (char*)d_ws;
    float* xbuf = (float*)(ws);                         // [0, S)       x  (f32)
    bfu*   xbf  = (bfu*)  (ws + S);                     // [S, 1.5S)    x  (bf16)
    bfu*   big  = (bfu*)  (ws + 3 * S / 2);             // [1.5S, 3.5S) qkv/ff1 (bf16)
    bfu*   qh   = (bfu*)  (ws + 7 * S / 2);             // [3.5S, 4S)
    bfu*   kh   = (bfu*)  (ws + 4 * S);                 // [4S, 4.5S)
    bfu*   vT   = (bfu*)  (ws + 9 * S / 2);             // [4.5S, 5S)
    bfu*   ctxb = (bfu*)  (ws + 5 * S);                 // [5S, 5.5S)
    float* ybuf = (float*)(ws + 11 * S / 2);            // [5.5S, 6.5S) y (f32)

    char*  wtop    = ws + 13 * S / 2;                   // bf16 weights + pooled outs
    bfu*   Wqkv_bf = (bfu*)(wtop);                      // 2*768*256
    bfu*   Wo_bf   = Wqkv_bf + 2 * 3 * DD * DD;         // 2*256*256
    bfu*   W1_bf   = Wo_bf   + 2 * DD * DD;             // 2*1024*256
    bfu*   W2_bf   = W1_bf   + 2 * FFD * DD;            // 2*256*1024
    float* uout    = (float*)(W2_bf + 2 * DD * FFD);
    float* iout    = uout + (size_t)BB * 512;

    const dim3 blk256(256), blk128(128), blk32(32);
    const int elem_blocks = M;                          // M*D/256 = M blocks
    const int ln_blocks   = M / 8;
    const dim3 gemm_grid_qkv(3 * DD / 64, M / 64);
    const dim3 gemm_grid_d  (DD / 64,     M / 64);
    const dim3 gemm_grid_ff (FFD / 64,    M / 64);

    // 0) one-time weight conversions to bf16
    k_f2bf<<<(2 * 3 * DD * DD) / 256, blk256, 0, stream>>>(Wqkv, Wqkv_bf);
    k_f2bf<<<(2 * DD * DD)     / 256, blk256, 0, stream>>>(Wo,   Wo_bf);
    k_f2bf<<<(2 * FFD * DD)    / 256, blk256, 0, stream>>>(W1,   W1_bf);
    k_f2bf<<<(2 * DD * FFD)    / 256, blk256, 0, stream>>>(W2,   W2_bf);

    // 1) embeddings
    k_embed<<<elem_blocks, blk256, 0, stream>>>(attr, attr_inds, attr_emb, ind_emb,
                                                xbuf, xbf);

    // 2) transformer layers
    for (int l = 0; l < 2; ++l) {
        k_gemm<<<gemm_grid_qkv, blk128, 0, stream>>>(
            xbf, Wqkv_bf + (size_t)l * 3 * DD * DD, bqkv + (size_t)l * 3 * DD,
            big, 3 * DD, DD, /*bias|bf16out*/5);
        k_repack<<<elem_blocks, blk256, 0, stream>>>(big, qh, kh, vT);
        k_attn<<<BB * NH * (LL / 16), blk32, 0, stream>>>(qh, kh, vT, attr_lens, ctxb);
        k_gemm<<<gemm_grid_d, blk128, 0, stream>>>(
            ctxb, Wo_bf + (size_t)l * DD * DD, bo + (size_t)l * DD,
            ybuf, DD, DD, /*bias*/1);
        k_add_ln<<<ln_blocks, blk256, 0, stream>>>(
            xbuf, ybuf, ln1_s + (size_t)l * DD, ln1_b + (size_t)l * DD, xbuf, xbf);
        k_gemm<<<gemm_grid_ff, blk128, 0, stream>>>(
            xbf, W1_bf + (size_t)l * FFD * DD, b1 + (size_t)l * FFD,
            big, FFD, DD, /*bias|relu|bf16out*/7);
        k_gemm<<<gemm_grid_d, blk128, 0, stream>>>(
            big, W2_bf + (size_t)l * DD * FFD, b2 + (size_t)l * DD,
            ybuf, DD, FFD, /*bias*/1);
        k_add_ln<<<ln_blocks, blk256, 0, stream>>>(
            xbuf, ybuf, ln2_s + (size_t)l * DD, ln2_b + (size_t)l * DD, xbuf, xbf);
    }

    // 3) pooling + concat
    k_pool<<<BB, blk256, 0, stream>>>(xbuf, attr_tf, attr_feat, lens_u, lens_i,
                                      user_ids, item_ids, uemb, iemb,
                                      feat_w, feat_b, uout, iout);

    // 4) logits + mask + new_targets
    k_logits<<<(BB * NT) / 8, blk256, 0, stream>>>(pos_t, pos_l, neg_t, neg_l,
                                                   oeu, oei, uout, iout,
                                                   (float*)d_out);
}
